// Network_64261300682896
// MI455X (gfx1250) — matrix operand order repaired and checked by reading, not compiled, not run on hardware
//
#include <hip/hip_runtime.h>
#include <math.h>

// ---------------------------------------------------------------------------
// Types for CDNA5 WMMA (wave32): bf16 A/B fragments, f32 accumulator.
// ---------------------------------------------------------------------------
typedef __attribute__((ext_vector_type(16))) __bf16 bf16x16;
typedef __attribute__((ext_vector_type(8)))  float  f32x8;

union FragU { bf16x16 v; uint4 q[2]; };
union Pack8 { unsigned short u[8]; uint4 q; };

__device__ __forceinline__ unsigned short f32_to_bf16(float f) {
  union { float f; unsigned int u; } v; v.f = f;
  unsigned int u = v.u;
  unsigned int lsb = (u >> 16) & 1u;
  u += 0x7fffu + lsb;                 // round-to-nearest-even
  return (unsigned short)(u >> 16);
}

// ---------------------------------------------------------------------------
// Weight repack: fp32 [O][256][3][3] -> bf16 [O][9][256] (K-contiguous per tap)
// ---------------------------------------------------------------------------
__global__ void wt_to_bf16_k(const float* __restrict__ w,
                             unsigned short* __restrict__ o, int O) {
  int t = blockIdx.x * blockDim.x + threadIdx.x;
  if (t >= O * 2304) return;
  int oo = t / 2304;
  int r  = t - oo * 2304;
  int i  = r / 9;
  int kk = r - i * 9;
  o[((size_t)oo * 9 + kk) * 256 + i] = f32_to_bf16(w[t]);
}

// fp32 NCHW -> bf16 HWC (pos-major, channel-contiguous) per batch
__global__ void cast_chw_hwc_bf16_k(const float* __restrict__ in,
                                    unsigned short* __restrict__ out,
                                    int HW, int nb) {
  long long t = (long long)blockIdx.x * blockDim.x + threadIdx.x;
  long long tot = (long long)nb * 256 * HW;
  if (t >= tot) return;
  long long b = t / (256LL * HW);
  long long r = t - b * 256LL * HW;
  int c   = (int)(r / HW);
  int pos = (int)(r - (long long)c * HW);
  out[(b * HW + pos) * 256 + c] = f32_to_bf16(in[t]);
}

// ---------------------------------------------------------------------------
// Implicit-GEMM conv3x3 SAME, Cin=256, via v_wmma_f32_16x16x32_bf16.
// Activations bf16 HWC [pos][256]; weights bf16 [O][9][256].
// Block: 256 threads = 8 waves. Tile: M=256 (32 rows/wave -> 2 A-frags),
// N=64 (4 B-frags shared across M) => 8 WMMAs per K-step.
// Double-buffered LDS, software-pipelined, all-register staging (no spills):
// rows >= Cout are zeroed in LDS once, staging is exec-masked for them.
// mode 0: bf16 HWC output with bias+ReLU (tower layers, Cout==256)
// mode 1: fp32 position-major output [pos][Cout] with bias (head layers)
// ---------------------------------------------------------------------------
__global__ __launch_bounds__(256)
void conv3x3_wmma_k(const unsigned short* __restrict__ act, long long act_bs,
                    const unsigned short* __restrict__ wt,
                    const float* __restrict__ bias,
                    void* __restrict__ outv, long long out_bs,
                    int H, int W, int Cout, int mode) {
  const int HW = H * W;
  const int p0 = blockIdx.x * 64;
  const int o0 = blockIdx.y * 256;
  const unsigned short* ab = act + (long long)blockIdx.z * act_bs;

  __shared__ alignas(16) unsigned short As[2][256 * 32];  // weights: [row][K=32]
  __shared__ alignas(16) unsigned short Bs[2][64 * 48];   // im2col:  [pos=64][K pad 48]

  const int tid  = threadIdx.x;
  const int wave = tid >> 5;
  const int lane = tid & 31;
  const int mloc = lane & 15;   // A row / B col / D col within fragment
  const int hh   = lane >> 4;   // K-half selector

  // A-staging: thread stages one full 32-channel row (64 B, 4 x b128)
  const bool aval   = (o0 + tid) < Cout;
  const size_t awoff = (size_t)(o0 + tid) * 9 * 256;

  // Rows >= Cout are zero for the whole kernel: zero them once in both buffers.
  if (!aval) {
    uint4 z = {0u, 0u, 0u, 0u};
    uint4* a0 = (uint4*)(As[0] + tid * 32);
    uint4* a1 = (uint4*)(As[1] + tid * 32);
    a0[0] = z; a0[1] = z; a0[2] = z; a0[3] = z;
    a1[0] = z; a1[1] = z; a1[2] = z; a1[3] = z;
  }

  // B-staging decomposition: thread -> (pos j, 8-channel chunk)
  const int jstage = tid >> 2;        // 0..63
  const int cstage = (tid & 3) * 8;   // 0,8,16,24
  const int posj   = p0 + jstage;
  const bool pval  = posj < HW;
  const int yj = pval ? (posj / W) : 0;
  const int xj = pval ? (posj - yj * W) : 0;

  // 72 K-steps: it = kk*8 + cb; K-block = (tap kk, channel block cb*32)
  auto stage_load = [&](int it, uint4& a0, uint4& a1, uint4& a2, uint4& a3, uint4& bq) {
    const int kk = it >> 3;
    const int c0 = (it & 7) * 32;
    if (aval) {
      const uint4* src = (const uint4*)(wt + awoff + (size_t)kk * 256 + c0);
      a0 = src[0]; a1 = src[1]; a2 = src[2]; a3 = src[3];
    }
    uint4 z = {0u, 0u, 0u, 0u};
    bq = z;
    if (pval) {
      const int sy = yj + (kk / 3) - 1;
      const int sx = xj + (kk - 3 * (kk / 3)) - 1;
      if (sy >= 0 && sy < H && sx >= 0 && sx < W)
        bq = *(const uint4*)(ab + ((size_t)(sy * W + sx) * 256) + c0 + cstage);
    }
  };
  auto stage_store = [&](int buf, const uint4& a0, const uint4& a1,
                         const uint4& a2, const uint4& a3, const uint4& bq) {
    if (aval) {
      uint4* ad = (uint4*)(As[buf] + tid * 32);
      ad[0] = a0; ad[1] = a1; ad[2] = a2; ad[3] = a3;
    }
    *(uint4*)(Bs[buf] + jstage * 48 + cstage) = bq;
  };

  f32x8 acc[2][4] = {};

  uint4 aq0 = {}, aq1 = {}, aq2 = {}, aq3 = {}, bq = {};
  stage_load(0, aq0, aq1, aq2, aq3, bq);
  stage_store(0, aq0, aq1, aq2, aq3, bq);
  __syncthreads();

  for (int it = 0; it < 72; ++it) {
    const int buf = it & 1;
    const bool more = (it + 1) < 72;
    // 1) issue next step's global loads (hide latency under compute)
    if (more) stage_load(it + 1, aq0, aq1, aq2, aq3, bq);
    // prefetch the step-after-next weight chunk toward cache
    __builtin_prefetch(wt + awoff + (size_t)((it + 2) >> 3) * 256 + ((it + 2) & 7) * 32, 0, 1);

    // 2) compute current step: fragments are contiguous K-runs -> b128 LDS loads
    FragU af[2];
    #pragma unroll
    for (int mh = 0; mh < 2; ++mh) {
      const unsigned short* ap = As[buf] + (wave * 32 + mh * 16 + mloc) * 32 + 8 * hh;
      af[mh].q[0] = *(const uint4*)ap;          // K = 8*hh .. +7
      af[mh].q[1] = *(const uint4*)(ap + 16);   // K = 16+8*hh .. +7
    }
    #pragma unroll
    for (int nf = 0; nf < 4; ++nf) {
      FragU bf;
      const unsigned short* bp = Bs[buf] + (size_t)(nf * 16 + mloc) * 48 + 8 * hh;
      bf.q[0] = *(const uint4*)bp;              // K = 8*hh .. +7
      bf.q[1] = *(const uint4*)(bp + 16);
      #pragma unroll
      for (int mh = 0; mh < 2; ++mh)
        acc[mh][nf] = __builtin_amdgcn_wmma_f32_16x16x32_bf16(
            false, af[mh].v, false, bf.v, (short)0, acc[mh][nf], false, false);
    }

    // 3) commit staged data to the other buffer, 4) single barrier
    if (more) stage_store(1 - buf, aq0, aq1, aq2, aq3, bq);
    __syncthreads();
  }

  // ---- epilogue ----
  if (mode == 0) {
    // bf16 HWC out, bias+ReLU; lane's 8 rows are contiguous channels
    unsigned short* ob = (unsigned short*)outv + (long long)blockIdx.z * out_bs;
    #pragma unroll
    for (int mh = 0; mh < 2; ++mh) {
      const int obase = o0 + wave * 32 + mh * 16 + 8 * hh;
      #pragma unroll
      for (int nf = 0; nf < 4; ++nf) {
        const int pos = p0 + nf * 16 + mloc;
        if (pos >= HW) continue;
        Pack8 pk;
        #pragma unroll
        for (int r = 0; r < 8; ++r) {
          float v = acc[mh][nf][r] + bias[obase + r];
          v = v > 0.f ? v : 0.f;
          pk.u[r] = f32_to_bf16(v);
        }
        *(uint4*)(ob + (size_t)pos * 256 + obase) = pk.q;
      }
    }
  } else {
    float* ob = (float*)outv + (long long)blockIdx.z * out_bs;
    #pragma unroll
    for (int mh = 0; mh < 2; ++mh) {
      #pragma unroll
      for (int nf = 0; nf < 4; ++nf) {
        const int pos = p0 + nf * 16 + mloc;
        if (pos >= HW) continue;
        #pragma unroll
        for (int r = 0; r < 8; ++r) {
          const int o = o0 + wave * 32 + mh * 16 + r + 8 * hh;
          if (o >= Cout) continue;
          ob[(size_t)pos * Cout + o] = acc[mh][nf][r] + bias[o];
        }
      }
    }
  }
}

// ---------------------------------------------------------------------------
// Anchor / target / loss helpers
// ---------------------------------------------------------------------------
__device__ __forceinline__ float iou_pp1(float ax0, float ay0, float ax1, float ay1,
                                         float gx0, float gy0, float gx1, float gy1) {
  float ab = (ax1 - ax0 + 1.f) * (ay1 - ay0 + 1.f);
  float ag = (gx1 - gx0 + 1.f) * (gy1 - gy0 + 1.f);
  float iw = fminf(ax1, gx1) - fmaxf(ax0, gx0) + 1.f; iw = iw > 0.f ? iw : 0.f;
  float ih = fminf(ay1, gy1) - fmaxf(ay0, gy0) + 1.f; ih = ih > 0.f ? ih : 0.f;
  float inter = iw * ih;
  return inter / (ab + ag - inter);
}

__device__ __forceinline__ void bbox_tf(const float a[4], const float* g, float out[4]) {
  float aw = a[2] - a[0] + 1.f, ah = a[3] - a[1] + 1.f;
  float acx = a[0] + 0.5f * (aw - 1.f), acy = a[1] + 0.5f * (ah - 1.f);
  float gw = g[2] - g[0] + 1.f, gh = g[3] - g[1] + 1.f;
  float gcx = g[0] + 0.5f * (gw - 1.f), gcy = g[1] + 0.5f * (gh - 1.f);
  out[0] = (gcx - acx) / aw; out[1] = (gcy - acy) / ah;
  out[2] = logf(gw / aw);    out[3] = logf(gh / ah);
}

__device__ __forceinline__ void bbox_tf_inv(const float a[4], const float* d, float out[4]) {
  float w = a[2] - a[0] + 1.f, h = a[3] - a[1] + 1.f;
  float cx = a[0] + 0.5f * (w - 1.f), cy = a[1] + 0.5f * (h - 1.f);
  const float mx = 4.135166556742356f;  // log(1000/16)
  float pw = expf(fminf(d[2], mx)) * w;
  float ph = expf(fminf(d[3], mx)) * h;
  float pcx = d[0] * w + cx, pcy = d[1] * h + cy;
  out[0] = pcx - 0.5f * (pw - 1.f); out[1] = pcy - 0.5f * (ph - 1.f);
  out[2] = pcx + 0.5f * (pw - 1.f); out[3] = pcy + 0.5f * (ph - 1.f);
}

struct LvlTab { int h[5]; int w[5]; int s[5]; int hwoff[5]; };

__global__ void make_anchors_k(LvlTab T, float* __restrict__ anchors, int A) {
  int i = blockIdx.x * blockDim.x + threadIdx.x;
  if (i >= A) return;
  int l = 0;
  for (int q = 1; q < 5; ++q) if (i >= 3 * T.hwoff[q]) l = q;
  int pin  = i - 3 * T.hwoff[l];
  int cell = pin / 3, a = pin - cell * 3;
  int y = cell / T.w[l], x = cell - y * T.w[l];
  float r  = (float)(a + 1);                 // ratios {1,2,3}
  float ws = roundf(sqrtf(16.f / r));
  float hs = roundf(ws * r);
  const float ctr = 1.5f;
  float x0 = ctr - 0.5f * (ws - 1.f), y0 = ctr - 0.5f * (hs - 1.f);
  float x1 = ctr + 0.5f * (ws - 1.f), y1 = ctr + 0.5f * (hs - 1.f);
  float aw = x1 - x0 + 1.f, ah = y1 - y0 + 1.f;
  float cx = x0 + 0.5f * (aw - 1.f), cy = y0 + 0.5f * (ah - 1.f);
  float st = (float)T.s[l];
  float sw = aw * st, sh = ah * st;
  float shx = x * st, shy = y * st;
  anchors[i * 4 + 0] = cx - 0.5f * (sw - 1.f) + shx;
  anchors[i * 4 + 1] = cy - 0.5f * (sh - 1.f) + shy;
  anchors[i * 4 + 2] = cx + 0.5f * (sw - 1.f) + shx;
  anchors[i * 4 + 3] = cy + 0.5f * (sh - 1.f) + shy;
}

__global__ void anchor_target_k(const float* __restrict__ gt, const float* __restrict__ iminfo,
                                const float* __restrict__ anchors,
                                float* __restrict__ labels, float* __restrict__ tgt,
                                int n, int A, int G) {
  int t = blockIdx.x * blockDim.x + threadIdx.x;
  if (t >= n * A) return;
  int b = t / A, i = t - b * A;
  float cnt = iminfo[b * 6 + 5];
  float a4[4] = {anchors[i * 4], anchors[i * 4 + 1], anchors[i * 4 + 2], anchors[i * 4 + 3]};
  float best = -3.0e38f, sec = -3.0e38f; int bi = 0, si = 0;
  for (int g = 0; g < G; ++g) {
    const float* gg = gt + ((size_t)b * G + g) * 5;
    float ov = -1.0f;
    if (((float)g < cnt) && (gg[4] != -1.0f))
      ov = iou_pp1(a4[0], a4[1], a4[2], a4[3], gg[0], gg[1], gg[2], gg[3]);
    if (ov > best)      { sec = best; si = bi; best = ov; bi = g; }
    else if (ov > sec)  { sec = ov; si = g; }
  }
  float la = best >= 0.5f ? 1.f : (best < 0.4f ? 0.f : -1.f);
  float lb = sec  >= 0.5f ? 1.f : (sec  < 0.4f ? 0.f : -1.f);
  float ign = lb - ((la == 0.f) ? 1.f : 0.f) * ((lb != 0.f) ? 1.f : 0.f);
  size_t lbase = (size_t)b * 2 * A + 2 * (size_t)i;
  labels[lbase]     = la;
  labels[lbase + 1] = ign;
  float o4[4];
  float* tb = tgt + lbase * 4;
  bbox_tf(a4, gt + ((size_t)b * G + bi) * 5, o4);
  tb[0] = o4[0]; tb[1] = o4[1]; tb[2] = o4[2]; tb[3] = o4[3];
  bbox_tf(a4, gt + ((size_t)b * G + si) * 5, o4);
  tb[4] = o4[0]; tb[5] = o4[1]; tb[6] = o4[2]; tb[7] = o4[3];
}

__global__ void iou_target_k(const float* __restrict__ gt, const float* __restrict__ iminfo,
                             const float* __restrict__ anchors, const float* __restrict__ bpred,
                             float* __restrict__ iout, int n, int A, int G) {
  int t = blockIdx.x * blockDim.x + threadIdx.x;
  if (t >= n * A) return;
  int b = t / A, i = t - b * A;
  float cnt = iminfo[b * 6 + 5];
  float a4[4] = {anchors[i * 4], anchors[i * 4 + 1], anchors[i * 4 + 2], anchors[i * 4 + 3]};
  const float* off = bpred + ((size_t)b * A + i) * 8;
  float dt0[4], dt1[4];
  bbox_tf_inv(a4, off, dt0);
  bbox_tf_inv(a4, off + 4, dt1);
  float best = -3.0e38f; int i1 = 0;
  for (int g = 0; g < G; ++g) {
    const float* gg = gt + ((size_t)b * G + g) * 5;
    float k = (((float)g < cnt) && (gg[4] != -1.0f)) ? 1.f : 0.f;
    float ov = iou_pp1(dt0[0], dt0[1], dt0[2], dt0[3], gg[0], gg[1], gg[2], gg[3]) * k;
    if (ov > best) { best = ov; i1 = g; }
  }
  float bv = 0.f;
  for (int g = 0; g < G; ++g) {
    if (g == i1) continue;
    const float* gg = gt + ((size_t)b * G + g) * 5;
    float k = (((float)g < cnt) && (gg[4] != -1.0f)) ? 1.f : 0.f;
    float ov = iou_pp1(dt1[0], dt1[1], dt1[2], dt1[3], gg[0], gg[1], gg[2], gg[3]) * k;
    bv = fmaxf(bv, ov);
  }
  iout[(size_t)b * A * 2 + (size_t)i * 2]     = best;
  iout[(size_t)b * A * 2 + (size_t)i * 2 + 1] = bv;
}

// focal loss + smooth-L1, per row of labels (n, 2A)
__global__ void cls_bbox_loss_k(const float* __restrict__ cls_raw, const float* __restrict__ labels,
                                const float* __restrict__ bpred, const float* __restrict__ tgt,
                                float* __restrict__ acc, int n, int A2) {
  int t = blockIdx.x * blockDim.x + threadIdx.x;
  if (t >= n * A2) return;
  float lab = labels[t];
  float raw = cls_raw[t];
  float maskf = (lab != -1.f) ? 1.f : 0.f;
  float lp, l1mp;                           // log(sigmoid), log(1-sigmoid)
  if (raw >= 0.f) { float e = expf(-raw); lp = -log1pf(e); l1mp = -raw - log1pf(e); }
  else            { float e = expf(raw);  lp = raw - log1pf(e); l1mp = -log1pf(e); }
  float p  = 1.f / (1.f + expf(-raw));
  float oh = (maskf != 0.f && lab == 1.f) ? 1.f : 0.f;
  float om = 1.f - p;
  float pos = om * om * oh * lp;
  float neg = p * p * (1.f - oh) * l1mp;
  atomicAdd(&acc[0], -(0.25f * pos + 0.75f * neg) * maskf);
  if (lab > 0.f) {
    atomicAdd(&acc[1], 1.f);
    float s = 0.f;
    #pragma unroll
    for (int c = 0; c < 4; ++c) {
      float d  = bpred[(size_t)t * 4 + c] - tgt[(size_t)t * 4 + c];
      float ad = fabsf(d);
      s += (ad < (1.f / 9.f)) ? (4.5f * d * d) : (ad - 1.f / 18.f);
    }
    atomicAdd(&acc[2], s);
    atomicAdd(&acc[3], 1.f);
  }
}

__global__ void iou_num_loss_k(const float* __restrict__ iou_raw, const float* __restrict__ num_raw,
                               const float* __restrict__ iout, const float* __restrict__ labels,
                               float* __restrict__ acc, int n, int A) {
  int t = blockIdx.x * blockDim.x + threadIdx.x;
  if (t >= n * A) return;
  int b = t / A, i = t - b * A;
  size_t lbase = (size_t)b * 2 * A + 2 * (size_t)i;
  float la = labels[lbase], lg = labels[lbase + 1];
  size_t pbase = (size_t)b * A * 2 + (size_t)i * 2;
  float isum = 0.f, icnt = 0.f;
  if (la > 0.f) { isum += fabsf(iou_raw[pbase]     - iout[pbase]);     icnt += 1.f; }
  if (lg > 0.f) { isum += fabsf(iou_raw[pbase + 1] - iout[pbase + 1]); icnt += 1.f; }
  if (icnt > 0.f) { atomicAdd(&acc[4], isum); atomicAdd(&acc[5], icnt); }
  int nl = ((la > 0.f) ? 1 : 0) + ((lg > 0.f) ? 1 : 0) - 1;
  if (nl != -1) {
    float s0 = num_raw[pbase], s1 = num_raw[pbase + 1];
    float m  = fmaxf(s0, s1);
    float lse = m + logf(expf(s0 - m) + expf(s1 - m));
    float lpv = ((nl == 0) ? s0 : s1) - lse;
    atomicAdd(&acc[6], -lpv);
    atomicAdd(&acc[7], 1.f);
  }
}

__global__ void zero_acc_k(float* acc) { if (threadIdx.x < 16) acc[threadIdx.x] = 0.f; }

__global__ void finalize_k(const float* __restrict__ acc, float* __restrict__ out) {
  if (threadIdx.x == 0) {
    out[0] = acc[0] / fmaxf(acc[1], 1.f);
    out[1] = 2.f * acc[2] / fmaxf(acc[3], 1.f);
    out[2] = 2.f * acc[4] / fmaxf(acc[5], 1.f);
    out[3] = acc[6] / fmaxf(acc[7], 1.f);
  }
}

// ---------------------------------------------------------------------------
// Host orchestration
// ---------------------------------------------------------------------------
extern "C" void kernel_launch(void* const* d_in, const int* in_sizes, int n_in,
                              void* d_out, int out_size, void* d_ws, size_t ws_size,
                              hipStream_t stream) {
  (void)in_sizes; (void)n_in; (void)out_size; (void)ws_size;
  const int n = 2, G = 32;
  const int Hs[5] = {6, 12, 24, 48, 96};
  const int Ws[5] = {10, 20, 40, 80, 160};
  const int Ss[5] = {128, 64, 32, 16, 8};
  int hw[5], hwoff[5]; int cum = 0;
  for (int l = 0; l < 5; ++l) { hwoff[l] = cum; hw[l] = Hs[l] * Ws[l]; cum += hw[l]; }
  const int HWT = cum;        // 20460
  const int A = 3 * HWT;      // 61380
  const int HWMAX = hw[4];

  char* base = (char*)d_ws;
  size_t off = 0;
  auto carve = [&](size_t bytes) -> char* {
    char* p = base + off;
    off += (bytes + 255) & ~(size_t)255;
    return p;
  };
  unsigned short* wbf_cls   = (unsigned short*)carve((size_t)4 * 256 * 9 * 256 * 2);
  unsigned short* wbf_box   = (unsigned short*)carve((size_t)4 * 256 * 9 * 256 * 2);
  unsigned short* wbf_score = (unsigned short*)carve((size_t)6 * 9 * 256 * 2);
  unsigned short* wbf_bpred = (unsigned short*)carve((size_t)24 * 9 * 256 * 2);
  unsigned short* wbf_iou   = (unsigned short*)carve((size_t)6 * 9 * 256 * 2);
  unsigned short* wbf_num   = (unsigned short*)carve((size_t)6 * 9 * 256 * 2);
  unsigned short* feat_bf   = (unsigned short*)carve((size_t)n * 256 * HWT * 2);
  unsigned short* tb0       = (unsigned short*)carve((size_t)n * 256 * HWMAX * 2);
  unsigned short* tb1       = (unsigned short*)carve((size_t)n * 256 * HWMAX * 2);
  float* cls_raw   = (float*)carve((size_t)n * HWT * 6 * 4);
  float* num_raw   = (float*)carve((size_t)n * HWT * 6 * 4);
  float* bpred_raw = (float*)carve((size_t)n * HWT * 24 * 4);
  float* iou_raw   = (float*)carve((size_t)n * HWT * 6 * 4);
  float* anchors   = (float*)carve((size_t)A * 4 * 4);
  float* labels    = (float*)carve((size_t)n * 2 * A * 4);
  float* tgt       = (float*)carve((size_t)n * 2 * A * 4 * 4);
  float* iout      = (float*)carve((size_t)n * A * 2 * 4);
  float* acc       = (float*)carve(64);

  // --- weight repacks (fp32 -> bf16 [O][9][256]) ---
  auto wconv = [&](const float* w, unsigned short* o, int O) {
    int tot = O * 2304;
    wt_to_bf16_k<<<(tot + 255) / 256, 256, 0, stream>>>(w, o, O);
  };
  wconv((const float*)d_in[7],  wbf_cls,   4 * 256);
  wconv((const float*)d_in[9],  wbf_box,   4 * 256);
  wconv((const float*)d_in[11], wbf_score, 6);
  wconv((const float*)d_in[13], wbf_bpred, 24);
  wconv((const float*)d_in[15], wbf_iou,   6);
  wconv((const float*)d_in[17], wbf_num,   6);

  // --- feature transpose-casts: fp32 NCHW -> bf16 HWC ---
  size_t fboff[5];
  { size_t c = 0; for (int l = 0; l < 5; ++l) { fboff[l] = c; c += (size_t)n * 256 * hw[l]; } }
  for (int l = 0; l < 5; ++l) {
    long long tot = (long long)n * 256 * hw[l];
    cast_chw_hwc_bf16_k<<<(int)((tot + 255) / 256), 256, 0, stream>>>(
        (const float*)d_in[l], feat_bf + fboff[l], hw[l], n);
  }

  auto conv = [&](const unsigned short* act, long long abs_, const unsigned short* wt,
                  const float* bias, void* outp, long long obs, int H, int W, int Cout, int mode) {
    dim3 grid((H * W + 63) / 64, (Cout + 255) / 256, n);
    conv3x3_wmma_k<<<grid, 256, 0, stream>>>(act, abs_, wt, bias, outp, obs, H, W, Cout, mode);
  };

  const float* cls_b   = (const float*)d_in[8];
  const float* box_b   = (const float*)d_in[10];
  const float* score_b = (const float*)d_in[12];
  const float* bpred_b = (const float*)d_in[14];
  const float* iou_b   = (const float*)d_in[16];
  const float* num_b   = (const float*)d_in[18];
  const size_t WLSTR = (size_t)256 * 9 * 256;

  for (int l = 0; l < 5; ++l) {
    const int H = Hs[l], W = Ws[l];
    const long long bs = (long long)256 * hw[l];
    const unsigned short* fin = feat_bf + fboff[l];
    // cls tower + heads
    const unsigned short* cur = fin;
    for (int i = 0; i < 4; ++i) {
      unsigned short* o = (i & 1) ? tb1 : tb0;
      conv(cur, bs, wbf_cls + (size_t)i * WLSTR, cls_b + i * 256, o, bs, H, W, 256, 0);
      cur = o;
    }
    conv(cur, bs, wbf_score, score_b, cls_raw + (size_t)hwoff[l] * 6, (long long)HWT * 6, H, W, 6, 1);
    conv(cur, bs, wbf_num,   num_b,   num_raw + (size_t)hwoff[l] * 6, (long long)HWT * 6, H, W, 6, 1);
    // box tower + heads
    cur = fin;
    for (int i = 0; i < 4; ++i) {
      unsigned short* o = (i & 1) ? tb1 : tb0;
      conv(cur, bs, wbf_box + (size_t)i * WLSTR, box_b + i * 256, o, bs, H, W, 256, 0);
      cur = o;
    }
    conv(cur, bs, wbf_bpred, bpred_b, bpred_raw + (size_t)hwoff[l] * 24, (long long)HWT * 24, H, W, 24, 1);
    conv(cur, bs, wbf_iou,   iou_b,   iou_raw   + (size_t)hwoff[l] * 6,  (long long)HWT * 6,  H, W, 6, 1);
  }

  LvlTab T;
  for (int l = 0; l < 5; ++l) { T.h[l] = Hs[l]; T.w[l] = Ws[l]; T.s[l] = Ss[l]; T.hwoff[l] = hwoff[l]; }
  make_anchors_k<<<(A + 255) / 256, 256, 0, stream>>>(T, anchors, A);
  zero_acc_k<<<1, 32, 0, stream>>>(acc);

  const float* gt     = (const float*)d_in[5];
  const float* iminfo = (const float*)d_in[6];
  anchor_target_k<<<(n * A + 255) / 256, 256, 0, stream>>>(gt, iminfo, anchors, labels, tgt, n, A, G);
  iou_target_k<<<(n * A + 255) / 256, 256, 0, stream>>>(gt, iminfo, anchors, bpred_raw, iout, n, A, G);
  cls_bbox_loss_k<<<(n * 2 * A + 255) / 256, 256, 0, stream>>>(cls_raw, labels, bpred_raw, tgt, acc, n, 2 * A);
  iou_num_loss_k<<<(n * A + 255) / 256, 256, 0, stream>>>(iou_raw, num_raw, iout, labels, acc, n, A);
  finalize_k<<<1, 32, 0, stream>>>(acc, (float*)d_out);
}
